// EulerResNetSeqModel_45775761441106
// MI455X (gfx1250) — matrix operand order repaired and checked by reading, not compiled, not run on hardware
//
#include <hip/hip_runtime.h>
#include <hip/hip_bf16.h>
#include <math.h>

typedef __attribute__((ext_vector_type(16))) _Float16 v16h;
typedef __attribute__((ext_vector_type(8)))  _Float16 v8h;
typedef __attribute__((ext_vector_type(8)))  float    v8f;

#define BATCH   65536
#define TSTEPS  7
#define WAVES   4          // waves per block, 16 batch rows per wave
#define SA_STR  360        // halves; 720B rows -> 180 dwords -> conflict-free
#define SB_STR  200        // halves; 400B rows -> 100 dwords -> conflict-free

__device__ __forceinline__ v8f zero8() {
  v8f z = {0.f,0.f,0.f,0.f,0.f,0.f,0.f,0.f};
  return z;
}

// Branch-free GELU (tanh form) using v_exp_f32 / v_rcp_f32 transcendentals.
// tanh(y) = 1 - 2/(exp(2y)+1); exp2 overflow -> rcp(inf)=0 -> t=1 (correct tail).
__device__ __forceinline__ float gelu_f(float x) {
  float y = 0.7978845608028654f * (x + 0.044715f * x * x * x);
  float e = __builtin_amdgcn_exp2f(y * 2.885390081777927f);   // exp(2y)
  float t = 1.f - 2.f * __builtin_amdgcn_rcpf(e + 1.f);
  return 0.5f * x * (1.f + t);
}

__device__ __forceinline__ v8f wmma16(v16h a, v16h b, v8f c) {
  return __builtin_amdgcn_wmma_f32_16x16x32_f16(false, a, false, b, (short)0, c, false, false);
}

// A fragment (16x32 f16) from LDS row-major buffer, stride in halves.
// Layout: lanes 0-15: M=lane, K = 0..15; lanes 16-31: K = 16..31 (matches B fragment).
__device__ __forceinline__ v16h load_afrag(const _Float16* buf, int stride, int kb, int lane) {
  int m  = lane & 15;
  int k0 = kb * 32 + ((lane < 16) ? 0 : 16);
  const _Float16* p = buf + m * stride + k0;
  v8h lo = *(const v8h*)(p);
  v8h hi = *(const v8h*)(p + 8);
  v16h r;
#pragma unroll
  for (int e = 0; e < 8; e++) { r[e] = lo[e]; r[e + 8] = hi[e]; }
  return r;
}

// B fragment (32x16 f16) from global transposed weights Wt[n][k], K padded to Kp.
__device__ __forceinline__ v16h load_bfrag(const _Float16* wt, int Kp, int nb, int kb, int lane) {
  int n  = nb * 16 + (lane & 15);
  int k0 = kb * 32 + ((lane < 16) ? 0 : 16);
  return *(const v16h*)(wt + (size_t)n * Kp + k0);
}

// Row-wise LN stats in fragment domain: ps/psq are per-VGPR-row partial sums,
// reduced across the 16 lanes of each wave half (row M = mb + i).
__device__ __forceinline__ void rowstats(float* ps, float* psq, float invF,
                                         float* mean, float* rs) {
#pragma unroll
  for (int off = 1; off < 16; off <<= 1)
#pragma unroll
    for (int i = 0; i < 8; i++) {
      ps[i]  += __shfl_xor(ps[i],  off, 32);
      psq[i] += __shfl_xor(psq[i], off, 32);
    }
#pragma unroll
  for (int i = 0; i < 8; i++) {
    float m = ps[i] * invF;
    mean[i] = m;
    rs[i]   = rsqrtf(psq[i] * invF - m * m + 1e-5f);
  }
}

// Head: partial[i] = sum_n gelu(z@W1 + b1)[row, n] * w2[n], reduced across 16-lane halves.
__device__ __forceinline__ void head_eval(const _Float16* sA, int lane,
                                          const _Float16* wt, const float* b1,
                                          const float* w2, int NB, float* partial) {
#pragma unroll
  for (int i = 0; i < 8; i++) partial[i] = 0.f;
  for (int nb = 0; nb < NB; nb++) {
    v8f acc = zero8();
    for (int kb = 0; kb < 11; kb++)               // K = 352
      acc = wmma16(load_afrag(sA, SA_STR, kb, lane), load_bfrag(wt, 352, nb, kb, lane), acc);
    int n = nb * 16 + (lane & 15);
    float bb = b1[n], ww = w2[n];
#pragma unroll
    for (int i = 0; i < 8; i++) partial[i] += gelu_f(acc[i] + bb) * ww;
  }
#pragma unroll
  for (int off = 1; off < 16; off <<= 1)
#pragma unroll
    for (int i = 0; i < 8; i++) partial[i] += __shfl_xor(partial[i], off, 32);
}

// Weight prep: f32 (K x N) row-major -> f16 transposed (N x Kp), zero-padded K.
__global__ void convert_w(const float* __restrict__ src, _Float16* __restrict__ dst,
                          int K, int N, int Kp) {
  int i = blockIdx.x * blockDim.x + threadIdx.x;
  if (i >= N * Kp) return;
  int n = i / Kp, k = i % Kp;
  dst[i] = (k < K) ? (_Float16)src[(size_t)k * N + n] : (_Float16)0.f;
}

__global__ __launch_bounds__(WAVES * 32)
void fused_forward(const float* __restrict__ x_num, const float* __restrict__ x_static,
                   const int* __restrict__ x_cat, const float* __restrict__ cat_emb,
                   const float* __restrict__ day_emb,
                   const _Float16* __restrict__ xtw, const float* __restrict__ xproj_b,
                   const float* __restrict__ h0,
                   const float* __restrict__ bln_g, const float* __restrict__ bln_b,
                   const _Float16* __restrict__ fc1t, const float* __restrict__ fc1_b,
                   const _Float16* __restrict__ fc2t, const float* __restrict__ fc2_b,
                   const float* __restrict__ spln_g, const float* __restrict__ spln_b,
                   const _Float16* __restrict__ sp1t, const float* __restrict__ sp1_b,
                   const _Float16* __restrict__ sp2t, const float* __restrict__ sp2_b,
                   const float* __restrict__ base_w, const float* __restrict__ base_b,
                   const float* __restrict__ hln_g, const float* __restrict__ hln_b,
                   const _Float16* __restrict__ hm1t, const float* __restrict__ hm1_b,
                   const float* __restrict__ hm2_w, const float* __restrict__ hm2_b,
                   const _Float16* __restrict__ hq1t, const float* __restrict__ hq1_b,
                   const float* __restrict__ hq2_w, const float* __restrict__ hq2_b,
                   const _Float16* __restrict__ hn1t, const float* __restrict__ hn1_b,
                   const float* __restrict__ hn2_w, const float* __restrict__ hn2_b,
                   float* __restrict__ out) {
  __shared__ _Float16 sAall[WAVES][16 * SA_STR];
  __shared__ _Float16 sBall[WAVES][16 * SB_STR];
  __shared__ float    sBase[WAVES][16];

  const int lane = threadIdx.x & 31;
  const int w    = threadIdx.x >> 5;
  _Float16* sA = sAall[w];
  _Float16* sB = sBall[w];
  const int r0 = blockIdx.x * (WAVES * 16) + w * 16;
  const int n15 = lane & 15;
  const int mb  = (lane < 16) ? 0 : 8;

  // ---------------- Phase S: static path ----------------
  {
    int row = lane >> 1, part = lane & 1;
    const float* xs = x_static + (size_t)(r0 + row) * 64 + part * 32;
    float s = 0.f, sq = 0.f, bs = 0.f;
    for (int j = 0; j < 32; j++) {
      float v = xs[j];
      s += v; sq += v * v; bs += v * base_w[part * 32 + j];
    }
    float s2  = s  + __shfl_xor(s, 1, 32);
    float sq2 = sq + __shfl_xor(sq, 1, 32);
    float bs2 = bs + __shfl_xor(bs, 1, 32);
    float mean = s2 / 64.f;
    float var  = sq2 / 64.f - mean * mean;
    float rs   = rsqrtf(var + 1e-5f);
    for (int j = 0; j < 32; j++) {
      float v = xs[j];
      sA[row * SA_STR + part * 32 + j] =
          (_Float16)((v - mean) * rs * spln_g[part * 32 + j] + spln_b[part * 32 + j]);
    }
    if (part == 0) sBase[w][row] = bs2 + base_b[0];
  }
  // zero K-pad columns 296..319 once (xproj GEMM reads K=320)
  for (int i = lane; i < 16 * 24; i += 32)
    sA[(i / 24) * SA_STR + 296 + (i % 24)] = (_Float16)0.f;

  // sp1: K=64, N=128 -> gelu -> sB
  for (int nb = 0; nb < 8; nb++) {
    v8f acc = zero8();
    for (int kb = 0; kb < 2; kb++)
      acc = wmma16(load_afrag(sA, SA_STR, kb, lane), load_bfrag(sp1t, 64, nb, kb, lane), acc);
    int n = nb * 16 + n15;
    float b1 = sp1_b[n];
#pragma unroll
    for (int i = 0; i < 8; i++)
      sB[(mb + i) * SB_STR + n] = (_Float16)gelu_f(acc[i] + b1);
  }
  // sp2: K=128, N=64 -> s64 fragments (kept in registers)
  v8f s64frag[4];
#pragma unroll
  for (int nb = 0; nb < 4; nb++) {
    v8f acc = zero8();
    for (int kb = 0; kb < 4; kb++)
      acc = wmma16(load_afrag(sB, SB_STR, kb, lane), load_bfrag(sp2t, 128, nb, kb, lane), acc);
    float b2 = sp2_b[nb * 16 + n15];
#pragma unroll
    for (int i = 0; i < 8; i++) acc[i] += b2;
    s64frag[nb] = acc;
  }

  // ---------------- recurrent state in f32 fragments ----------------
  v8f hfrag[6], hsum[6], hmax[6];
#pragma unroll
  for (int nb = 0; nb < 6; nb++) {
    float hv = h0[nb * 16 + n15];
#pragma unroll
    for (int i = 0; i < 8; i++) {
      hfrag[nb][i] = hv; hsum[nb][i] = 0.f; hmax[nb][i] = -3.4e38f;
    }
  }

  // ---------------- Phase R: 7 Euler steps ----------------
#pragma unroll 1
  for (int t = 0; t < TSTEPS; t++) {
    // build x_t (f16) in sA: [x_num(32) | emb(256) | day(8) | pad(24)]
#pragma unroll 1
    for (int row = 0; row < 16; row++) {
      float v = x_num[(((size_t)(r0 + row)) * TSTEPS + t) * 32 + lane];
      sA[row * SA_STR + lane] = (_Float16)v;
    }
#pragma unroll 1
    for (int i = lane; i < 256; i += 32) {
      int row = i >> 4, f = i & 15;
      int idx = x_cat[(((size_t)(r0 + row)) * TSTEPS + t) * 16 + f];
      const float4* e4 = (const float4*)(cat_emb + ((size_t)f * 101 + idx) * 16);
      float4 a0 = e4[0], a1 = e4[1], a2 = e4[2], a3 = e4[3];
      v8h lo, hi;
      lo[0]=(_Float16)a0.x; lo[1]=(_Float16)a0.y; lo[2]=(_Float16)a0.z; lo[3]=(_Float16)a0.w;
      lo[4]=(_Float16)a1.x; lo[5]=(_Float16)a1.y; lo[6]=(_Float16)a1.z; lo[7]=(_Float16)a1.w;
      hi[0]=(_Float16)a2.x; hi[1]=(_Float16)a2.y; hi[2]=(_Float16)a2.z; hi[3]=(_Float16)a2.w;
      hi[4]=(_Float16)a3.x; hi[5]=(_Float16)a3.y; hi[6]=(_Float16)a3.z; hi[7]=(_Float16)a3.w;
      _Float16* dst = sA + row * SA_STR + 32 + f * 16;
      *(v8h*)dst = lo; *(v8h*)(dst + 8) = hi;
    }
    for (int i = lane; i < 128; i += 32)
      sA[(i >> 3) * SA_STR + 288 + (i & 7)] = (_Float16)day_emb[(1 + t) * 8 + (i & 7)];

    // xproj: K=320, N=96 (keep all 6 accs; scatter only after K loop since dst aliases src)
    v8f acc6[6];
#pragma unroll
    for (int nb = 0; nb < 6; nb++) acc6[nb] = zero8();
    for (int kb = 0; kb < 10; kb++) {
      v16h a = load_afrag(sA, SA_STR, kb, lane);
#pragma unroll
      for (int nb = 0; nb < 6; nb++)
        acc6[nb] = wmma16(a, load_bfrag(xtw, 320, nb, kb, lane), acc6[nb]);
    }
#pragma unroll
    for (int nb = 0; nb < 6; nb++) {
      float bb = xproj_b[nb * 16 + n15];
#pragma unroll
      for (int i = 0; i < 8; i++) acc6[nb][i] += bb;
    }

    // fused LayerNorm over z = [h | xt] (F=192), stats computed in f32 fragment domain
    float ps[8], psq[8], mean[8], rsv[8];
#pragma unroll
    for (int i = 0; i < 8; i++) { ps[i] = 0.f; psq[i] = 0.f; }
#pragma unroll
    for (int nb = 0; nb < 6; nb++)
#pragma unroll
      for (int i = 0; i < 8; i++) {
        float hv = hfrag[nb][i], xv = acc6[nb][i];
        ps[i]  += hv + xv;
        psq[i] += hv * hv + xv * xv;
      }
    rowstats(ps, psq, 1.f / 192.f, mean, rsv);

    // scatter normalized z (f16) into sA cols 0..191
#pragma unroll
    for (int nb = 0; nb < 6; nb++) {
      int c0 = nb * 16 + n15;
      float g0 = bln_g[c0],      b0 = bln_b[c0];
      float g1 = bln_g[96 + c0], b1 = bln_b[96 + c0];
#pragma unroll
      for (int i = 0; i < 8; i++) {
        sA[(mb + i) * SA_STR + c0]      = (_Float16)((hfrag[nb][i] - mean[i]) * rsv[i] * g0 + b0);
        sA[(mb + i) * SA_STR + 96 + c0] = (_Float16)((acc6[nb][i]  - mean[i]) * rsv[i] * g1 + b1);
      }
    }

    // fc1: K=192, N=192 -> gelu -> sB
    for (int nb = 0; nb < 12; nb++) {
      v8f acc = zero8();
      for (int kb = 0; kb < 6; kb++)
        acc = wmma16(load_afrag(sA, SA_STR, kb, lane), load_bfrag(fc1t, 192, nb, kb, lane), acc);
      int n = nb * 16 + n15;
      float b1 = fc1_b[n];
#pragma unroll
      for (int i = 0; i < 8; i++)
        sB[(mb + i) * SB_STR + n] = (_Float16)gelu_f(acc[i] + b1);
    }
    // fc2: K=192, N=96 -> dh; Euler update + running mean/max (all in f32 regs)
#pragma unroll
    for (int nb = 0; nb < 6; nb++) {
      v8f acc = zero8();
      for (int kb = 0; kb < 6; kb++)
        acc = wmma16(load_afrag(sB, SB_STR, kb, lane), load_bfrag(fc2t, 192, nb, kb, lane), acc);
      float b2 = fc2_b[nb * 16 + n15];
#pragma unroll
      for (int i = 0; i < 8; i++) {
        float h = hfrag[nb][i] + (acc[i] + b2);   // DT = 1
        hfrag[nb][i] = h;
        hsum[nb][i] += h;
        hmax[nb][i]  = fmaxf(hmax[nb][i], h);
      }
    }
  }

  // ---------------- Phase F: z = [h_last | h_mean | h_max | s64], fused LN (F=352) ----------------
  {
    const float inv7 = 1.f / 7.f;
    float ps[8], psq[8], mean[8], rsv[8];
#pragma unroll
    for (int i = 0; i < 8; i++) { ps[i] = 0.f; psq[i] = 0.f; }
#pragma unroll
    for (int nb = 0; nb < 6; nb++)
#pragma unroll
      for (int i = 0; i < 8; i++) {
        float a = hfrag[nb][i], m = hsum[nb][i] * inv7, x = hmax[nb][i];
        ps[i]  += a + m + x;
        psq[i] += a * a + m * m + x * x;
      }
#pragma unroll
    for (int nb = 0; nb < 4; nb++)
#pragma unroll
      for (int i = 0; i < 8; i++) {
        float v = s64frag[nb][i];
        ps[i] += v; psq[i] += v * v;
      }
    rowstats(ps, psq, 1.f / 352.f, mean, rsv);

#pragma unroll
    for (int nb = 0; nb < 6; nb++) {
      int c0 = nb * 16 + n15;
      float g0 = hln_g[c0],       b0 = hln_b[c0];
      float g1 = hln_g[96 + c0],  b1 = hln_b[96 + c0];
      float g2 = hln_g[192 + c0], b2 = hln_b[192 + c0];
#pragma unroll
      for (int i = 0; i < 8; i++) {
        sA[(mb + i) * SA_STR + c0]       = (_Float16)((hfrag[nb][i] - mean[i]) * rsv[i] * g0 + b0);
        sA[(mb + i) * SA_STR + 96 + c0]  = (_Float16)((hsum[nb][i] * inv7 - mean[i]) * rsv[i] * g1 + b1);
        sA[(mb + i) * SA_STR + 192 + c0] = (_Float16)((hmax[nb][i] - mean[i]) * rsv[i] * g2 + b2);
      }
    }
#pragma unroll
    for (int nb = 0; nb < 4; nb++) {
      int c0 = 288 + nb * 16 + n15;
      float g3 = hln_g[c0], b3 = hln_b[c0];
#pragma unroll
      for (int i = 0; i < 8; i++)
        sA[(mb + i) * SA_STR + c0] = (_Float16)((s64frag[nb][i] - mean[i]) * rsv[i] * g3 + b3);
    }
  }

  float pm[8], pq[8], pn[8];
  head_eval(sA, lane, hm1t, hm1_b, hm2_w, 12, pm);
  head_eval(sA, lane, hq1t, hq1_b, hq2_w, 12, pq);
  head_eval(sA, lane, hn1t, hn1_b, hn2_w,  6, pn);

  int j = n15;
  if (j < 8) {
    // select partial[j] without dynamic register indexing
    float vm = pm[0], vq = pq[0], vn = pn[0];
#pragma unroll
    for (int i = 1; i < 8; i++) {
      vm = (j == i) ? pm[i] : vm;
      vq = (j == i) ? pq[i] : vq;
      vn = (j == i) ? pn[i] : vn;
    }
    int row = mb + j;
    size_t rg = (size_t)(r0 + row);
    float bs = sBase[w][row];
    out[rg]                     = bs + vm + hm2_b[0];
    out[(size_t)BATCH + rg]     = bs + vq + hq2_b[0];
    out[2 * (size_t)BATCH + rg] = vn + hn2_b[0];
  }
}

extern "C" void kernel_launch(void* const* d_in, const int* in_sizes, int n_in,
                              void* d_out, int out_size, void* d_ws, size_t ws_size,
                              hipStream_t stream) {
  const float* x_num    = (const float*)d_in[0];
  const float* x_static = (const float*)d_in[1];
  const int*   x_cat    = (const int*)  d_in[2];
  const float* cat_emb  = (const float*)d_in[3];
  const float* day_emb  = (const float*)d_in[4];
  const float* xproj_w  = (const float*)d_in[5];
  const float* xproj_b  = (const float*)d_in[6];
  const float* h0       = (const float*)d_in[7];
  const float* bln_g    = (const float*)d_in[8];
  const float* bln_b    = (const float*)d_in[9];
  const float* fc1_w    = (const float*)d_in[10];
  const float* fc1_b    = (const float*)d_in[11];
  const float* fc2_w    = (const float*)d_in[12];
  const float* fc2_b    = (const float*)d_in[13];
  const float* spln_g   = (const float*)d_in[14];
  const float* spln_b   = (const float*)d_in[15];
  const float* sp1_w    = (const float*)d_in[16];
  const float* sp1_b    = (const float*)d_in[17];
  const float* sp2_w    = (const float*)d_in[18];
  const float* sp2_b    = (const float*)d_in[19];
  const float* base_w   = (const float*)d_in[20];
  const float* base_b   = (const float*)d_in[21];
  const float* hln_g    = (const float*)d_in[22];
  const float* hln_b    = (const float*)d_in[23];
  const float* hm1_w    = (const float*)d_in[24];
  const float* hm1_b    = (const float*)d_in[25];
  const float* hm2_w    = (const float*)d_in[26];
  const float* hm2_b    = (const float*)d_in[27];
  const float* hq1_w    = (const float*)d_in[28];
  const float* hq1_b    = (const float*)d_in[29];
  const float* hq2_w    = (const float*)d_in[30];
  const float* hq2_b    = (const float*)d_in[31];
  const float* hn1_w    = (const float*)d_in[32];
  const float* hn1_b    = (const float*)d_in[33];
  const float* hn2_w    = (const float*)d_in[34];
  const float* hn2_b    = (const float*)d_in[35];

  // workspace: transposed f16 weights (offsets in halves)
  _Float16* wsp  = (_Float16*)d_ws;
  _Float16* xtw  = wsp;                 // 96  x 320
  _Float16* fc1t = wsp + 30720;         // 192 x 192
  _Float16* fc2t = wsp + 67584;         // 96  x 192
  _Float16* sp1t = wsp + 86016;         // 128 x 64
  _Float16* sp2t = wsp + 94208;         // 64  x 128
  _Float16* hm1t = wsp + 102400;        // 192 x 352
  _Float16* hq1t = wsp + 169984;        // 192 x 352
  _Float16* hn1t = wsp + 237568;        // 96  x 352

  auto conv = [&](const float* s, _Float16* d, int K, int N, int Kp) {
    int tot = N * Kp;
    convert_w<<<(tot + 255) / 256, 256, 0, stream>>>(s, d, K, N, Kp);
  };
  conv(xproj_w, xtw, 296,  96, 320);
  conv(fc1_w,  fc1t, 192, 192, 192);
  conv(fc2_w,  fc2t, 192,  96, 192);
  conv(sp1_w,  sp1t,  64, 128,  64);
  conv(sp2_w,  sp2t, 128,  64, 128);
  conv(hm1_w,  hm1t, 352, 192, 352);
  conv(hq1_w,  hq1t, 352, 192, 352);
  conv(hn1_w,  hn1t, 352,  96, 352);

  fused_forward<<<BATCH / (WAVES * 16), WAVES * 32, 0, stream>>>(
      x_num, x_static, x_cat, cat_emb, day_emb,
      xtw, xproj_b, h0, bln_g, bln_b,
      fc1t, fc1_b, fc2t, fc2_b,
      spln_g, spln_b, sp1t, sp1_b, sp2t, sp2_b,
      base_w, base_b, hln_g, hln_b,
      hm1t, hm1_b, hm2_w, hm2_b,
      hq1t, hq1_b, hq2_w, hq2_b,
      hn1t, hn1_b, hn2_w, hn2_b,
      (float*)d_out);
}
